// EdgeFeatsConvMultNN_2327872274905
// MI455X (gfx1250) — compile-verified, matrix operand order checked by reading, not run on hardware
//
#include <hip/hip_runtime.h>
#include <hip/hip_bf16.h>

// ---------------------------------------------------------------------------
// Types for gfx1250 WMMA (wave32): v_wmma_f32_16x16x32_bf16
// ---------------------------------------------------------------------------
typedef __attribute__((ext_vector_type(16))) __bf16 v16bf;
typedef __attribute__((ext_vector_type(8)))  float  v8f;

// fp32 -> bf16, round-to-nearest-even
__device__ __forceinline__ unsigned short f2bf(float f) {
  unsigned int u = __float_as_uint(f);
  u += 0x7FFFu + ((u >> 16) & 1u);
  return (unsigned short)(u >> 16);
}

// Byte offset of element (rowOrCol, kw) inside one 1KB 16x32 bf16 fragment.
// A-matrix 16x32 layout (ISA 7.12.2): lane = m + 16*((kw>>3)&1),
// half-index j = (kw&7) + 8*(kw>>4). B mirrors with n in place of m.
__device__ __forceinline__ int fragOff(int rc, int kw) {
  int hi = (kw >> 3) & 1;
  int j  = (kw & 7) + ((kw >> 4) << 3);
  return ((rc + (hi << 4)) << 5) + (j << 1);
}

__device__ __forceinline__ void st16(char* s, int base, int rc, int kw, unsigned short v) {
  *(unsigned short*)(s + base + fragOff(rc, kw)) = v;
}

// store four consecutive-K bf16 values (kw % 4 == 0) as one b64:
// kw%8 in {0,4} keeps j..j+3 inside one 8-group -> 8 contiguous aligned bytes
__device__ __forceinline__ void stQuad(char* s, int base, int row, int kw, float4 v) {
  unsigned long long q = (unsigned long long)f2bf(v.x)
                       | ((unsigned long long)f2bf(v.y) << 16)
                       | ((unsigned long long)f2bf(v.z) << 32)
                       | ((unsigned long long)f2bf(v.w) << 48);
  *(unsigned long long*)(s + base + fragOff(row, kw)) = q;
}

// one lane's 32B of a fragment (contiguous -> 2x ds_load_b128)
__device__ __forceinline__ v16bf ldFrag(const char* s, int base, int frag, int lane) {
  return *(const v16bf*)(s + base + (frag << 10) + (lane << 5));
}

__device__ __forceinline__ float mishf(float v) {
  float sp = (v > 20.0f) ? v : __logf(1.0f + __expf(v));
  return v * tanhf(sp);
}

__device__ __forceinline__ v8f wmma_bf16(v16bf a, v16bf b, v8f c) {
  // (neg_a, A, neg_b, B, c_mod, C, reuse_a, reuse_b)
  return __builtin_amdgcn_wmma_f32_16x16x32_bf16(false, a, false, b, (short)0, c, false, false);
}

// ---------------------------------------------------------------------------
// Pass A: out = x @ Wroot + bias  (16-node tiles, K=128, 8 waves x 16 cols)
// ---------------------------------------------------------------------------
#define R_WRB   0                 // Wroot swizzled: 4 kchunks * 8 ntiles * 1KB
#define R_AXB   32768             // x tile fragments: 4 * 1KB
#define R_LDS   36864

__global__ void __launch_bounds__(256)
root_kernel(const float* __restrict__ x, const float* __restrict__ Wr,
            const float* __restrict__ bias, float* __restrict__ out, int N) {
  extern __shared__ v8f smem_raw[];
  char* smem = (char*)smem_raw;
  const int tid  = threadIdx.x;
  const int lane = tid & 31;
  const int wave = tid >> 5;
  const int col  = (wave << 4) + (lane & 15);

  for (int i4 = tid; i4 < 128 * 32; i4 += 256) {      // float4 along n
    int k = i4 >> 5, n4 = (i4 & 31) << 2;
    float4 w = *(const float4*)(Wr + (k << 7) + n4);
    int fb = R_WRB + ((((k >> 5) << 3) + (n4 >> 4)) << 10);
    st16(smem, fb, (n4 + 0) & 15, k & 31, f2bf(w.x));
    st16(smem, fb, (n4 + 1) & 15, k & 31, f2bf(w.y));
    st16(smem, fb, (n4 + 2) & 15, k & 31, f2bf(w.z));
    st16(smem, fb, (n4 + 3) & 15, k & 31, f2bf(w.w));
  }
  const float bv = bias[col];
  __syncthreads();

  const int nTiles = (N + 15) >> 4;
  for (int t = blockIdx.x; t < nTiles; t += gridDim.x) {
    const int base = t << 4;
    const bool full = (base + 16 <= N);
    if (full) {
#pragma unroll
      for (int it = 0; it < 2; ++it) {
        int task = tid + (it << 8);        // 16 rows * 32 float4 cols
        int e = task >> 5;
        int c = (task & 31) << 2;
        float4 v = *(const float4*)(x + ((size_t)(base + e) << 7) + c);
        stQuad(smem, R_AXB + ((c >> 5) << 10), e, c & 31, v);
      }
    } else {
#pragma unroll
      for (int it = 0; it < 2; ++it) {
        int task = tid + (it << 8);
        int e = task >> 5;
        int c = (task & 31) << 2;
        float4 v = {0.f, 0.f, 0.f, 0.f};
        if (base + e < N) v = *(const float4*)(x + ((size_t)(base + e) << 7) + c);
        stQuad(smem, R_AXB + ((c >> 5) << 10), e, c & 31, v);
      }
    }
    __syncthreads();

    v8f acc = {0, 0, 0, 0, 0, 0, 0, 0};
#pragma unroll
    for (int kk = 0; kk < 4; ++kk) {
      v16bf a = ldFrag(smem, R_AXB, kk, lane);
      v16bf b = ldFrag(smem, R_WRB, (kk << 3) + wave, lane);
      acc = wmma_bf16(a, b, acc);
    }
    if (full) {
#pragma unroll
      for (int r = 0; r < 8; ++r) {
        int m = r + ((lane >> 4) << 3);
        out[((size_t)(base + m) << 7) + col] = acc[r] + bv;
      }
    } else {
#pragma unroll
      for (int r = 0; r < 8; ++r) {
        int m = r + ((lane >> 4) << 3);
        if (base + m < N) out[((size_t)(base + m) << 7) + col] = acc[r] + bv;
      }
    }
    __syncthreads();
  }
}

// ---------------------------------------------------------------------------
// Pass B: per-edge message GEMMs + atomic scatter-add
// LDS layout (dynamic, 156800 B -> 2 workgroups per WGP of the 320KB pool)
// ---------------------------------------------------------------------------
#define E_W1B   0                  // 8 kchunks * 8 ntiles * 1KB = 64KB
#define E_W2B   65536              // 4 * 8 * 1KB = 32KB
#define E_WEB   98304              // 5 * 8 * 1KB = 40KB (K padded 144->160)
#define E_A1B   139264             // concat(x_i, x_j): 8 * 1KB
#define E_A2B   147456             // mish(h1) bf16:    4 * 1KB
#define E_A3B   151552             // concat(attr,x_j): 5 * 1KB
#define E_IDXB  156672             // dst[16] ints, src[16] ints
#define E_LDS   156800

__global__ void __launch_bounds__(256)
edge_kernel(const float* __restrict__ x, const long long* __restrict__ ei,
            const float* __restrict__ eattr,
            const float* __restrict__ W1, const float* __restrict__ b1,
            const float* __restrict__ W2, const float* __restrict__ b2,
            const float* __restrict__ We, const float* __restrict__ be,
            float* __restrict__ out, int E) {
  extern __shared__ v8f smem_raw[];
  char* smem = (char*)smem_raw;
  const int tid  = threadIdx.x;
  const int lane = tid & 31;
  const int wave = tid >> 5;
  const int col  = (wave << 4) + (lane & 15);
  int* idxD = (int*)(smem + E_IDXB);
  int* idxS = (int*)(smem + E_IDXB + 64);

  // ---- one-time: weights -> LDS in B-fragment layout (bf16) ----
  for (int i4 = tid; i4 < 256 * 32; i4 += 256) {
    int k = i4 >> 5, n4 = (i4 & 31) << 2;
    float4 w = *(const float4*)(W1 + (k << 7) + n4);
    int fb = E_W1B + ((((k >> 5) << 3) + (n4 >> 4)) << 10);
    st16(smem, fb, (n4 + 0) & 15, k & 31, f2bf(w.x));
    st16(smem, fb, (n4 + 1) & 15, k & 31, f2bf(w.y));
    st16(smem, fb, (n4 + 2) & 15, k & 31, f2bf(w.z));
    st16(smem, fb, (n4 + 3) & 15, k & 31, f2bf(w.w));
  }
  for (int i4 = tid; i4 < 128 * 32; i4 += 256) {
    int k = i4 >> 5, n4 = (i4 & 31) << 2;
    float4 w = *(const float4*)(W2 + (k << 7) + n4);
    int fb = E_W2B + ((((k >> 5) << 3) + (n4 >> 4)) << 10);
    st16(smem, fb, (n4 + 0) & 15, k & 31, f2bf(w.x));
    st16(smem, fb, (n4 + 1) & 15, k & 31, f2bf(w.y));
    st16(smem, fb, (n4 + 2) & 15, k & 31, f2bf(w.z));
    st16(smem, fb, (n4 + 3) & 15, k & 31, f2bf(w.w));
  }
  for (int i4 = tid; i4 < 160 * 32; i4 += 256) {
    int k = i4 >> 5, n4 = (i4 & 31) << 2;
    float4 w = {0.f, 0.f, 0.f, 0.f};
    if (k < 144) w = *(const float4*)(We + (k << 7) + n4);   // zero-pad K 144..159
    int fb = E_WEB + ((((k >> 5) << 3) + (n4 >> 4)) << 10);
    st16(smem, fb, (n4 + 0) & 15, k & 31, f2bf(w.x));
    st16(smem, fb, (n4 + 1) & 15, k & 31, f2bf(w.y));
    st16(smem, fb, (n4 + 2) & 15, k & 31, f2bf(w.z));
    st16(smem, fb, (n4 + 3) & 15, k & 31, f2bf(w.w));
  }
  // zero A3 pad region (chunk 4, kw 16..31) once; never rewritten per tile
  { int row = tid >> 4, kw = 16 + (tid & 15);
    *(unsigned short*)(smem + E_A3B + (4 << 10) + fragOff(row, kw)) = 0; }

  const float b1v = b1[col], b2v = b2[col], bev = be[col];
  __syncthreads();

  const int nTiles = (E + 15) >> 4;
  for (int t = blockIdx.x; t < nTiles; t += gridDim.x) {
    const int base = t << 4;
    const bool full = (base + 16 <= E);

    if (tid < 16) {
      int e = base + tid;
      idxS[tid] = (e < E) ? (int)ei[e] : 0;              // src j
      idxD[tid] = (e < E) ? (int)ei[(size_t)E + e] : 0;  // dst i
    }
    __syncthreads();

    // ---- stage activations as bf16 A-fragments (float4 gathers) ----
#pragma unroll
    for (int it = 0; it < 2; ++it) {
      int task = tid + (it << 8);          // 16 edges * 32 float4 columns
      int e = task >> 5;
      int c = (task & 31) << 2;
      float4 vi = *(const float4*)(x + ((size_t)idxD[e] << 7) + c);  // x_i
      float4 vj = *(const float4*)(x + ((size_t)idxS[e] << 7) + c);  // x_j
      stQuad(smem, E_A1B + ((c >> 5) << 10), e, c & 31, vi);         // A1 K=c
      int k2 = 128 + c;
      stQuad(smem, E_A1B + ((k2 >> 5) << 10), e, k2 & 31, vj);       // A1 K=128+c
      int k3 = 16 + c;
      stQuad(smem, E_A3B + ((k3 >> 5) << 10), e, k3 & 31, vj);       // A3 K=16+c
    }
    if (tid < 64) {                        // edge_attr: 16 edges * 4 float4
      int e = tid >> 2;
      int c = (tid & 3) << 2;
      int eg = base + e;
      float4 a = {0.f, 0.f, 0.f, 0.f};
      if (eg < E) a = *(const float4*)(eattr + ((size_t)eg << 4) + c);
      stQuad(smem, E_A3B, e, c, a);
    }
    __syncthreads();

    // ---- GEMM1: h1 = mish(concat(x_i,x_j) @ W1 + b1) ----
    v8f acc = {0, 0, 0, 0, 0, 0, 0, 0};
#pragma unroll
    for (int kk = 0; kk < 8; ++kk) {
      v16bf a = ldFrag(smem, E_A1B, kk, lane);
      v16bf b = ldFrag(smem, E_W1B, (kk << 3) + wave, lane);
      acc = wmma_bf16(a, b, acc);
    }
    { // write mish(h1) as bf16 into A2 fragments (this wave's 16 K-columns)
      int kk2 = col >> 5, kw = col & 31;
      int j  = (kw & 7) + ((kw >> 4) << 3);
      int hi = (kw >> 3) & 1;
#pragma unroll
      for (int r = 0; r < 8; ++r) {
        float h = mishf(acc[r] + b1v);
        int tl = r + ((lane >> 4) << 3) + (hi << 4);
        *(unsigned short*)(smem + E_A2B + (kk2 << 10) + (tl << 5) + (j << 1)) = f2bf(h);
      }
    }
    __syncthreads();

    // ---- GEMM2: h_n = A2 @ W2 + b2 ;  GEMMe: h_e = mish(A3 @ We + be) ----
    v8f accn = {0, 0, 0, 0, 0, 0, 0, 0};
    v8f acce = {0, 0, 0, 0, 0, 0, 0, 0};
#pragma unroll
    for (int kk = 0; kk < 4; ++kk) {
      v16bf a = ldFrag(smem, E_A2B, kk, lane);
      v16bf b = ldFrag(smem, E_W2B, (kk << 3) + wave, lane);
      accn = wmma_bf16(a, b, accn);
    }
#pragma unroll
    for (int kk = 0; kk < 5; ++kk) {
      v16bf a = ldFrag(smem, E_A3B, kk, lane);
      v16bf b = ldFrag(smem, E_WEB, (kk << 3) + wave, lane);
      acce = wmma_bf16(a, b, acce);
    }

    // ---- msg = h_n * h_e ; scatter-add to out[dst] (L2-resident atomics) ----
    if (full) {
#pragma unroll
      for (int r = 0; r < 8; ++r) {
        int m = r + ((lane >> 4) << 3);
        float msg = (accn[r] + b2v) * mishf(acce[r] + bev);
        atomicAdd(out + ((size_t)idxD[m] << 7) + col, msg);
      }
    } else {
#pragma unroll
      for (int r = 0; r < 8; ++r) {
        int m = r + ((lane >> 4) << 3);
        if (base + m < E) {
          float msg = (accn[r] + b2v) * mishf(acce[r] + bev);
          atomicAdd(out + ((size_t)idxD[m] << 7) + col, msg);
        }
      }
    }
    __syncthreads();
  }
}

// ---------------------------------------------------------------------------
// Pass C/D: training-mode BatchNorm over N rows, 128 channels
// ---------------------------------------------------------------------------
__global__ void stats_kernel(const float* __restrict__ out, float* __restrict__ st, int N) {
  int col = threadIdx.x & 127;
  int row = (int)((blockIdx.x * blockDim.x + threadIdx.x) >> 7);
  int stride = (int)((gridDim.x * blockDim.x) >> 7);
  float s = 0.f, s2 = 0.f;
  for (int r = row; r < N; r += stride) {
    float v = out[(size_t)r * 128 + col];
    s += v; s2 += v * v;
  }
  atomicAdd(st + col, s);
  atomicAdd(st + 128 + col, s2);
}

__global__ void bn_kernel(float* __restrict__ out, const float* __restrict__ st,
                          const float* __restrict__ gamma, const float* __restrict__ beta, int N) {
  long long total = (long long)N * 128;
  long long i0 = (long long)blockIdx.x * blockDim.x + threadIdx.x;
  long long step = (long long)gridDim.x * blockDim.x;
  float invN = 1.0f / (float)N;
  for (long long i = i0; i < total; i += step) {
    int col = (int)(i & 127);
    float mean = st[col] * invN;
    float var  = st[128 + col] * invN - mean * mean;
    out[i] = (out[i] - mean) * rsqrtf(var + 1e-5f) * gamma[col] + beta[col];
  }
}

// ---------------------------------------------------------------------------
extern "C" void kernel_launch(void* const* d_in, const int* in_sizes, int n_in,
                              void* d_out, int out_size, void* d_ws, size_t ws_size,
                              hipStream_t stream) {
  const float*     x     = (const float*)d_in[0];
  const long long* ei    = (const long long*)d_in[1];   // int64 [2,E]
  const float*     eattr = (const float*)d_in[2];
  const float*     W1    = (const float*)d_in[3];
  const float*     b1    = (const float*)d_in[4];
  const float*     W2    = (const float*)d_in[5];
  const float*     b2    = (const float*)d_in[6];
  const float*     We    = (const float*)d_in[7];
  const float*     be    = (const float*)d_in[8];
  const float*     Wroot = (const float*)d_in[9];
  const float*     bias  = (const float*)d_in[10];
  const float*     gamma = (const float*)d_in[11];
  const float*     beta  = (const float*)d_in[12];
  float* out = (float*)d_out;
  float* st  = (float*)d_ws;

  const int N = in_sizes[0] / 128;
  const int E = in_sizes[2] / 16;

  (void)n_in; (void)out_size; (void)ws_size;
  (void)hipFuncSetAttribute((const void*)root_kernel,
                            hipFuncAttributeMaxDynamicSharedMemorySize, R_LDS);
  (void)hipFuncSetAttribute((const void*)edge_kernel,
                            hipFuncAttributeMaxDynamicSharedMemorySize, E_LDS);

  // Pass A: fully (re)initialize out = x @ Wroot + bias
  root_kernel<<<512, 256, R_LDS, stream>>>(x, Wroot, bias, out, N);

  // zero BN stats scratch (capturable)
  (void)hipMemsetAsync(st, 0, 256 * sizeof(float), stream);

  // Pass B: edge messages, atomic aggregation into out
  edge_kernel<<<2048, 256, E_LDS, stream>>>(x, ei, eattr, W1, b1, W2, b2, We, be, out, E);

  // Pass C/D: batch-norm
  stats_kernel<<<512, 256, 0, stream>>>(out, st, N);
  bn_kernel<<<1024, 256, 0, stream>>>(out, st, gamma, beta, N);
}
